// MOE_20486994002330
// MI455X (gfx1250) — compile-verified, matrix operand order checked by reading, not run on hardware
//
#include <hip/hip_runtime.h>

// ---------------- problem constants ----------------
#define DIMK     1024
#define INTER    2688
#define SHINTER  5376          // N_SHARE * INTER
#define NE       8
#define TOK      8192          // B*N
#define SLOT_CAP 16640         // 2*TOK + 8*32 padding, 32-aligned
#define NTILES32 (SLOT_CAP/32) // 520

typedef __attribute__((ext_vector_type(16))) __bf16 v16bf;
typedef __attribute__((ext_vector_type(8)))  float  v8f;

union FragU { v16bf v; uint4 q[2]; };

__device__ __forceinline__ unsigned short f2bfu(float f) {   // RNE f32->bf16
  union { float f; unsigned u; } v; v.f = f;
  return (unsigned short)((v.u + 0x7FFFu + ((v.u >> 16) & 1u)) >> 16);
}
__device__ __forceinline__ unsigned pk_bf16(float a, float b) {
  return (unsigned)f2bfu(a) | ((unsigned)f2bfu(b) << 16);
}

// WMMA fragment from bf16 memory: elems[0..7] at col kh*8, elems[8..15] at col 16+kh*8
__device__ __forceinline__ v16bf frag_bf16(const unsigned short* p, int kh) {
  FragU f;
  f.q[0] = *(const uint4*)(p + kh * 8);
  f.q[1] = *(const uint4*)(p + 16 + kh * 8);
  return f.v;
}
__device__ __forceinline__ v8f wmma_bf16(v16bf a, v16bf b, v8f c) {
  return __builtin_amdgcn_wmma_f32_16x16x32_bf16(false, a, false, b, (short)0, c, false, false);
}
__device__ __forceinline__ float silu(float x) {
  return x * __builtin_amdgcn_rcpf(1.0f + __expf(-x));   // fast v_rcp_f32
}

// ---------------- small setup kernels ----------------
__global__ void init_k(int* cnt, float* aux) {
  if (threadIdx.x < 16) cnt[threadIdx.x] = 0;
  if (threadIdx.x == 0) *aux = 0.0f;       // aux_loss = 0 (eval mode)
}

// generic f32 -> bf16 converter, 4 elems/thread (used for x and all weight staging)
__global__ void cvt_k(const float* __restrict__ src, unsigned short* __restrict__ dst) {
  size_t i = ((size_t)blockIdx.x * 256 + threadIdx.x) * 4;
  float4 f = *(const float4*)(src + i);
  uint2 o;
  o.x = pk_bf16(f.x, f.y);
  o.y = pk_bf16(f.z, f.w);
  *(uint2*)(dst + i) = o;
}

// one wave per token: logits -> softmax -> top-2 combine weights
__global__ void router_k(const float* __restrict__ x, const float* __restrict__ Wg,
                         float* __restrict__ we8, int* __restrict__ cnt) {
  int t = blockIdx.x, lane = threadIdx.x;
  const float* xr = x + (size_t)t * DIMK;
  float acc[NE] = {};
  for (int k = lane; k < DIMK; k += 32) {
    float xv = xr[k];
#pragma unroll
    for (int e = 0; e < NE; e++) acc[e] += xv * Wg[e * DIMK + k];
  }
#pragma unroll
  for (int m = 16; m >= 1; m >>= 1) {
#pragma unroll
    for (int e = 0; e < NE; e++) acc[e] += __shfl_xor(acc[e], m, 32);
  }
  if (lane == 0) {
    float mx = acc[0];
#pragma unroll
    for (int e = 1; e < NE; e++) mx = fmaxf(mx, acc[e]);
    float p[NE], s = 0.f;
#pragma unroll
    for (int e = 0; e < NE; e++) { p[e] = __expf(acc[e] - mx); s += p[e]; }
    int i1 = 0;
#pragma unroll
    for (int e = 1; e < NE; e++) if (p[e] > p[i1]) i1 = e;
    int i2 = (i1 == 0) ? 1 : 0;
#pragma unroll
    for (int e = 0; e < NE; e++) if (e != i1 && p[e] > p[i2]) i2 = e;
    float inv = __builtin_amdgcn_rcpf(s);
    float w1 = p[i1] * inv, w2 = p[i2] * inv;   // ROUTE_SCALE = 1.0
#pragma unroll
    for (int e = 0; e < NE; e++)
      we8[(size_t)t * NE + e] = (e == i1) ? w1 : ((e == i2) ? w2 : 0.0f);
    atomicAdd(&cnt[i1], 1);
    atomicAdd(&cnt[i2], 1);
  }
}

__global__ void offsets_k(const int* __restrict__ cnt, int* __restrict__ offs) {
  if (threadIdx.x == 0) {
    int o = 0;
#pragma unroll
    for (int e = 0; e < NE; e++) { offs[e] = o; o += ((cnt[e] + 31) >> 5) << 5; }
    offs[NE] = o;
  }
}

__global__ void slotinit_k(int* __restrict__ perm, float* __restrict__ wslot) {
  int i = blockIdx.x * 256 + threadIdx.x;
  if (i < SLOT_CAP) { perm[i] = 0; wslot[i] = 0.0f; }
}

__global__ void scatter_k(const float* __restrict__ we8, const int* __restrict__ offs,
                          int* __restrict__ cnt2, int* __restrict__ perm,
                          float* __restrict__ wslot) {
  int t = blockIdx.x * 256 + threadIdx.x;
#pragma unroll
  for (int e = 0; e < NE; e++) {
    float w = we8[(size_t)t * NE + e];
    if (w > 0.0f) {
      int pos = offs[e] + atomicAdd(&cnt2[e], 1);
      perm[pos] = t;
      wslot[pos] = w;
    }
  }
}

// ---------------- WMMA GEMM kernels (32M x 16N per wave, all-bf16 operands) ----------------
// gate: H = silu(X@B1^T) * (X@B3^T).  B1/B3 rows [n, ldk], pre-staged bf16.
// shared: grid (TOK/32, SHINTER/128).
__global__ __launch_bounds__(256) void gemm1_shared_k(
    const unsigned short* __restrict__ xb, const unsigned short* __restrict__ B1w,
    const unsigned short* __restrict__ B3w, unsigned short* __restrict__ Hs) {
  int wave = threadIdx.x >> 5, lane = threadIdx.x & 31;
  int r = lane & 15, kh = lane >> 4;
  int m0 = blockIdx.x * 32;
  int n0 = blockIdx.y * 128 + wave * 16;
  const unsigned short* A0 = xb + (size_t)(m0 + r) * DIMK;
  const unsigned short* A1 = xb + (size_t)(m0 + 16 + r) * DIMK;
  const unsigned short* B1 = B1w + (size_t)(n0 + r) * DIMK;
  const unsigned short* B3 = B3w + (size_t)(n0 + r) * DIMK;
  v8f c1a = {}, c1b = {}, c3a = {}, c3b = {};
  for (int kb = 0; kb < DIMK; kb += 32) {
    v16bf a0 = frag_bf16(A0 + kb, kh);
    v16bf a1 = frag_bf16(A1 + kb, kh);
    v16bf b1 = frag_bf16(B1 + kb, kh);
    v16bf b3 = frag_bf16(B3 + kb, kh);
    c1a = wmma_bf16(a0, b1, c1a);  c1b = wmma_bf16(a1, b1, c1b);
    c3a = wmma_bf16(a0, b3, c3a);  c3b = wmma_bf16(a1, b3, c3b);
  }
  int n = n0 + r;
#pragma unroll
  for (int v = 0; v < 8; v++) {
    int m = m0 + v + kh * 8;
    Hs[(size_t)m * SHINTER + n]        = f2bfu(silu(c1a[v]) * c3a[v]);
    Hs[(size_t)(m + 16) * SHINTER + n] = f2bfu(silu(c1b[v]) * c3b[v]);
  }
}

// shared down: out = Hs @ Ws2^T (bf16 staged), full f32 write.  grid (TOK/32, DIMK/128)
__global__ __launch_bounds__(256) void gemm2_shared_k(
    const unsigned short* __restrict__ Hs, const unsigned short* __restrict__ B2w,
    float* __restrict__ out) {
  int wave = threadIdx.x >> 5, lane = threadIdx.x & 31;
  int r = lane & 15, kh = lane >> 4;
  int m0 = blockIdx.x * 32;
  int n0 = blockIdx.y * 128 + wave * 16;
  const unsigned short* A0 = Hs + (size_t)(m0 + r) * SHINTER;
  const unsigned short* A1 = Hs + (size_t)(m0 + 16 + r) * SHINTER;
  const unsigned short* Bp = B2w + (size_t)(n0 + r) * SHINTER;
  v8f ca = {}, cb = {};
  for (int kb = 0; kb < SHINTER; kb += 32) {
    v16bf b = frag_bf16(Bp + kb, kh);
    ca = wmma_bf16(frag_bf16(A0 + kb, kh), b, ca);
    cb = wmma_bf16(frag_bf16(A1 + kb, kh), b, cb);
  }
  int n = n0 + r;
#pragma unroll
  for (int v = 0; v < 8; v++) {
    out[(size_t)(m0 + v + kh * 8) * DIMK + n]      = ca[v];
    out[(size_t)(m0 + 16 + v + kh * 8) * DIMK + n] = cb[v];
  }
}

// routed gate for ONE expert's slot region (others exit).  grid (NTILES32, INTER/128)
__global__ __launch_bounds__(256) void gemm1_routed_k(
    const unsigned short* __restrict__ xb, const unsigned short* __restrict__ B1w,
    const unsigned short* __restrict__ B3w, const int* __restrict__ offs,
    const int* __restrict__ perm, unsigned short* __restrict__ H, int e) {
  int m0 = blockIdx.x * 32;
  if (m0 < offs[e] || m0 >= offs[e + 1]) return;   // 32-aligned expert regions
  int wave = threadIdx.x >> 5, lane = threadIdx.x & 31;
  int r = lane & 15, kh = lane >> 4;
  int n0 = blockIdx.y * 128 + wave * 16;
  const unsigned short* A0 = xb + (size_t)perm[m0 + r] * DIMK;
  const unsigned short* A1 = xb + (size_t)perm[m0 + 16 + r] * DIMK;
  const unsigned short* B1 = B1w + (size_t)(n0 + r) * DIMK;
  const unsigned short* B3 = B3w + (size_t)(n0 + r) * DIMK;
  v8f c1a = {}, c1b = {}, c3a = {}, c3b = {};
  for (int kb = 0; kb < DIMK; kb += 32) {
    v16bf a0 = frag_bf16(A0 + kb, kh);
    v16bf a1 = frag_bf16(A1 + kb, kh);
    v16bf b1 = frag_bf16(B1 + kb, kh);
    v16bf b3 = frag_bf16(B3 + kb, kh);
    c1a = wmma_bf16(a0, b1, c1a);  c1b = wmma_bf16(a1, b1, c1b);
    c3a = wmma_bf16(a0, b3, c3a);  c3b = wmma_bf16(a1, b3, c3b);
  }
  int n = n0 + r;
#pragma unroll
  for (int v = 0; v < 8; v++) {
    int m = m0 + v + kh * 8;
    H[(size_t)m * INTER + n]        = f2bfu(silu(c1a[v]) * c3a[v]);
    H[(size_t)(m + 16) * INTER + n] = f2bfu(silu(c1b[v]) * c3b[v]);
  }
}

// routed down for ONE expert: out[tok] += (H @ W2[e]^T) * wslot.  grid (NTILES32, DIMK/128)
__global__ __launch_bounds__(256) void gemm2_routed_k(
    const unsigned short* __restrict__ H, const unsigned short* __restrict__ B2w,
    const int* __restrict__ offs, const int* __restrict__ perm,
    const float* __restrict__ wslot, float* __restrict__ out, int e) {
  int m0 = blockIdx.x * 32;
  if (m0 < offs[e] || m0 >= offs[e + 1]) return;
  int wave = threadIdx.x >> 5, lane = threadIdx.x & 31;
  int r = lane & 15, kh = lane >> 4;
  int n0 = blockIdx.y * 128 + wave * 16;
  const unsigned short* A0 = H + (size_t)(m0 + r) * INTER;
  const unsigned short* A1 = H + (size_t)(m0 + 16 + r) * INTER;
  const unsigned short* Bp = B2w + (size_t)(n0 + r) * INTER;
  v8f ca = {}, cb = {};
  for (int kb = 0; kb < INTER; kb += 32) {
    v16bf b = frag_bf16(Bp + kb, kh);
    ca = wmma_bf16(frag_bf16(A0 + kb, kh), b, ca);
    cb = wmma_bf16(frag_bf16(A1 + kb, kh), b, cb);
  }
  int n = n0 + r;
#pragma unroll
  for (int v = 0; v < 8; v++) {
    int row = m0 + v + kh * 8;
    unsafeAtomicAdd(&out[(size_t)perm[row] * DIMK + n], ca[v] * wslot[row]);
    unsafeAtomicAdd(&out[(size_t)perm[row + 16] * DIMK + n], cb[v] * wslot[row + 16]);
  }
}

// ---------------- launch ----------------
extern "C" void kernel_launch(void* const* d_in, const int* in_sizes, int n_in,
                              void* d_out, int out_size, void* d_ws, size_t ws_size,
                              hipStream_t stream) {
  const float* x   = (const float*)d_in[0];
  const float* Wg  = (const float*)d_in[1];
  const float* W1  = (const float*)d_in[2];
  const float* W2  = (const float*)d_in[3];
  const float* W3  = (const float*)d_in[4];
  const float* Ws1 = (const float*)d_in[5];
  const float* Ws2 = (const float*)d_in[6];
  const float* Ws3 = (const float*)d_in[7];
  float* out = (float*)d_out;

  char* ws = (char*)d_ws;
  size_t off = 0;
  unsigned short* xb = (unsigned short*)(ws + off); off += (size_t)TOK * DIMK * 2;
  float* we8         = (float*)(ws + off);          off += (size_t)TOK * NE * 4;
  int*   cnt         = (int*)(ws + off);            off += 64;   // cnt[0..7], cnt2[8..15]
  int*   offs        = (int*)(ws + off);            off += 64;
  int*   perm        = (int*)(ws + off);            off += (size_t)SLOT_CAP * 4;
  float* wslot       = (float*)(ws + off);          off += (size_t)SLOT_CAP * 4;
  off = (off + 255) & ~(size_t)255;
  unsigned short* wbA = (unsigned short*)(ws + off); off += (size_t)SHINTER * DIMK * 2;  // stage slot A
  unsigned short* wbB = (unsigned short*)(ws + off); off += (size_t)SHINTER * DIMK * 2;  // stage slot B
  unsigned short* H   = (unsigned short*)(ws + off); // max(SLOT_CAP*INTER, TOK*SHINTER) bf16

  // routing + activation conversion
  init_k<<<1, 64, 0, stream>>>(cnt, out + (size_t)TOK * DIMK);
  cvt_k<<<(TOK * DIMK) / 1024, 256, 0, stream>>>(x, xb);
  router_k<<<TOK, 32, 0, stream>>>(x, Wg, we8, cnt);
  offsets_k<<<1, 32, 0, stream>>>(cnt, offs);
  slotinit_k<<<(SLOT_CAP + 255) / 256, 256, 0, stream>>>(perm, wslot);
  scatter_k<<<TOK / 256, 256, 0, stream>>>(we8, offs, cnt + 8, perm, wslot);

  // shared expert (writes full `out`)
  cvt_k<<<(SHINTER * DIMK) / 1024, 256, 0, stream>>>(Ws1, wbA);
  cvt_k<<<(SHINTER * DIMK) / 1024, 256, 0, stream>>>(Ws3, wbB);
  gemm1_shared_k<<<dim3(TOK / 32, SHINTER / 128), 256, 0, stream>>>(xb, wbA, wbB, H);
  cvt_k<<<(DIMK * SHINTER) / 1024, 256, 0, stream>>>(Ws2, wbA);
  gemm2_shared_k<<<dim3(TOK / 32, DIMK / 128), 256, 0, stream>>>(H, wbA, out);

  // routed experts: stage each expert's weights in bf16, accumulate on top of `out`
  for (int e = 0; e < NE; e++) {
    cvt_k<<<(INTER * DIMK) / 1024, 256, 0, stream>>>(W1 + (size_t)e * INTER * DIMK, wbA);
    cvt_k<<<(INTER * DIMK) / 1024, 256, 0, stream>>>(W3 + (size_t)e * INTER * DIMK, wbB);
    gemm1_routed_k<<<dim3(NTILES32, INTER / 128), 256, 0, stream>>>(xb, wbA, wbB, offs, perm, H, e);
  }
  for (int e = 0; e < NE; e++) {
    cvt_k<<<(DIMK * INTER) / 1024, 256, 0, stream>>>(W2 + (size_t)e * DIMK * INTER, wbA);
    gemm2_routed_k<<<dim3(NTILES32, DIMK / 128), 256, 0, stream>>>(H, wbA, offs, perm, wslot, out, e);
  }
}
